// MiniSpinNet_64218351010432
// MI455X (gfx1250) — compile-verified, matrix operand order checked by reading, not run on hardware
//
#include <hip/hip_runtime.h>
#include <hip/hip_bf16.h>

// ---------------------------------------------------------------------------
// MiniSpinNet on gfx1250: per-keypoint LDS-resident pipeline, conv layers as
// V_WMMA_F32_16X16X32_F16 GEMMs. Activations staged channel-fastest into
// padded LDS grids so every A/B fragment is two aligned ds_load_b128 with no
// conditionals in the WMMA loop. Patch staging uses the CDNA5 async
// global->LDS engine (global_load_async_to_lds_b128 + s_wait_asynccnt).
// ---------------------------------------------------------------------------

typedef __attribute__((ext_vector_type(16))) _Float16 v16h;
typedef __attribute__((ext_vector_type(8)))  _Float16 v8h;
typedef __attribute__((ext_vector_type(8)))  float    v8f;

union V16H { v16h v; v8h r[2]; _Float16 h[16]; };
union V8F  { v8f  v; float f[8]; };

#define NPTS   20000
#define MKPT   512
#define PATCH  512
#define RADN   3
#define ELEN   7
#define AZIN   20
#define NVOX   420      // 3*7*20
#define NPOS2  140      // 7*20
#define VOXS   30
#define DESR   0.3f
#define VRAD   (0.8f/3.0f)
#define PI_F   3.14159265358979f
#define BN_EPS 1e-5f

static __device__ __forceinline__ int imin(int a, int b) { return a < b ? a : b; }

// ---------------------------------------------------------------------------
// Kernel 1: radius grouping. One wave per keypoint; ballot+prefix keeps the
// ascending-index order of the reference's sort-based selection.
// ---------------------------------------------------------------------------
__global__ __launch_bounds__(32)
void k_group(const float* __restrict__ pts, const float* __restrict__ kpts,
             float* __restrict__ dx) {
  int m = blockIdx.x, lane = threadIdx.x;
  float kx = kpts[m*3+0], ky = kpts[m*3+1], kz = kpts[m*3+2];
  __shared__ int sel[PATCH];
  int cnt = 0;
  for (int base = 0; base < NPTS; base += 32) {
    if (cnt >= PATCH) break;
    int i = base + lane;
    float ax = pts[i*3+0]-kx, ay = pts[i*3+1]-ky, az = pts[i*3+2]-kz;
    bool in = (ax*ax + ay*ay + az*az) < (DESR*DESR);
    unsigned mask = (unsigned)__ballot(in);
    int pre = __popc(mask & ((1u << lane) - 1u));
    if (in && (cnt + pre) < PATCH) sel[cnt + pre] = i;
    cnt += __popc(mask);
  }
  if (cnt > PATCH) cnt = PATCH;
  __syncthreads();
  for (int s = lane; s < PATCH; s += 32) {
    int idx = -1;
    if (s == 0)                       idx = (cnt > 0) ? sel[0] : (NPTS - 1);
    else if (s < cnt && s != PATCH-1) idx = sel[s];
    float ox = 0.f, oy = 0.f, oz = 0.f;
    if (idx >= 0) {
      ox = pts[idx*3+0]-kx; oy = pts[idx*3+1]-ky; oz = pts[idx*3+2]-kz;
    }
    dx[(m*PATCH + s)*3 + 0] = ox;
    dx[(m*PATCH + s)*3 + 1] = oy;
    dx[(m*PATCH + s)*3 + 2] = oz;
  }
}

// ---------------------------------------------------------------------------
// Kernel 2: covariance -> smallest eigenvector (closed form) -> Rodrigues
// frame; rotate dx in place, scale by 1/DES_R.
// ---------------------------------------------------------------------------
__global__ __launch_bounds__(32)
void k_frame(const float* __restrict__ kpts, float* __restrict__ patch) {
  int m = blockIdx.x, lane = threadIdx.x;
  float* dxp = patch + (size_t)m * PATCH * 3;
  float cxx=0,cxy=0,cxz=0,cyy=0,cyz=0,czz=0;
  for (int s = lane; s < PATCH; s += 32) {
    float x = dxp[s*3], y = dxp[s*3+1], z = dxp[s*3+2];
    cxx += x*x; cxy += x*y; cxz += x*z; cyy += y*y; cyz += y*z; czz += z*z;
  }
  for (int off = 16; off; off >>= 1) {
    cxx += __shfl_xor(cxx, off, 32); cxy += __shfl_xor(cxy, off, 32);
    cxz += __shfl_xor(cxz, off, 32); cyy += __shfl_xor(cyy, off, 32);
    cyz += __shfl_xor(cyz, off, 32); czz += __shfl_xor(czz, off, 32);
  }
  __shared__ float sR[9];
  if (lane == 0) {
    float a00=cxx,a01=cxy,a02=cxz,a11=cyy,a12=cyz,a22=czz;
    float zx=0.f, zy=0.f, zz=1.f;
    float p1 = a01*a01 + a02*a02 + a12*a12;
    if (p1 > 1e-20f) {
      float q  = (a00+a11+a22)*(1.f/3.f);
      float b00=a00-q, b11=a11-q, b22=a22-q;
      float p2 = b00*b00+b11*b11+b22*b22 + 2.f*p1;
      float p  = sqrtf(p2*(1.f/6.f));
      float iv = 1.f/p;
      float c00=b00*iv,c01=a01*iv,c02=a02*iv,c11=b11*iv,c12=a12*iv,c22=b22*iv;
      float det = c00*(c11*c22-c12*c12) - c01*(c01*c22-c12*c02)
                + c02*(c01*c12-c11*c02);
      float r = det*0.5f; r = r < -1.f ? -1.f : (r > 1.f ? 1.f : r);
      float phi = acosf(r)*(1.f/3.f);
      float lam = q + 2.f*p*cosf(phi + 2.09439510239f);   // smallest eigenvalue
      float r0x=a00-lam, r0y=a01,     r0z=a02;
      float r1x=a01,     r1y=a11-lam, r1z=a12;
      float r2x=a02,     r2y=a12,     r2z=a22-lam;
      float c0x=r0y*r1z-r0z*r1y, c0y=r0z*r1x-r0x*r1z, c0z=r0x*r1y-r0y*r1x;
      float c1x=r0y*r2z-r0z*r2y, c1y=r0z*r2x-r0x*r2z, c1z=r0x*r2y-r0y*r2x;
      float c2x=r1y*r2z-r1z*r2y, c2y=r1z*r2x-r1x*r2z, c2z=r1x*r2y-r1y*r2x;
      float n0=c0x*c0x+c0y*c0y+c0z*c0z, n1=c1x*c1x+c1y*c1y+c1z*c1z,
            n2=c2x*c2x+c2y*c2y+c2z*c2z;
      float vx=c0x, vy=c0y, vz=c0z, nn=n0;
      if (n1 > nn) { vx=c1x; vy=c1y; vz=c1z; nn=n1; }
      if (n2 > nn) { vx=c2x; vy=c2y; vz=c2z; nn=n2; }
      float inv = rsqrtf(nn + 1e-24f);
      zx = vx*inv; zy = vy*inv; zz = vz*inv;
    } else {
      zx = 0.f; zy = 0.f; zz = 1.f;
      if (a00 <= a11 && a00 <= a22) { zx = 1.f; zy = 0.f; zz = 0.f; }
      else if (a11 <= a22)          { zx = 0.f; zy = 1.f; zz = 0.f; }
    }
    float kx = kpts[m*3], ky = kpts[m*3+1], kz = kpts[m*3+2];
    if (zx*kx + zy*ky + zz*kz > 0.f) { zx=-zx; zy=-zy; zz=-zz; }
    float cx = zy, cy = -zx, cz = 0.f;           // z x ez
    float nz = sqrtf(zx*zx + zy*zy + zz*zz);
    float cv = zz / (nz + 1e-8f);
    cv = cv < -1.f+1e-6f ? -1.f+1e-6f : (cv > 1.f-1e-6f ? 1.f-1e-6f : cv);
    float th = acosf(cv), sth = sinf(th), oc = 1.f - cosf(th);
    float cn = sqrtf(cx*cx + cy*cy + cz*cz) + 1e-8f;
    float ux = cx/cn, uy = cy/cn, uz = cz/cn;
    float K[9]  = {0.f,-uz, uy,  uz, 0.f,-ux,  -uy, ux, 0.f};
    float K2[9];
    #pragma unroll
    for (int i = 0; i < 3; ++i)
      #pragma unroll
      for (int j = 0; j < 3; ++j)
        K2[i*3+j] = K[i*3]*K[j] + K[i*3+1]*K[3+j] + K[i*3+2]*K[6+j];
    #pragma unroll
    for (int i = 0; i < 9; ++i)
      sR[i] = (i%4==0 ? 1.f : 0.f) + sth*K[i] + oc*K2[i];
  }
  __syncthreads();
  float R00=sR[0],R01=sR[1],R02=sR[2],R10=sR[3],R11=sR[4],R12=sR[5],
        R20=sR[6],R21=sR[7],R22=sR[8];
  float ir = 1.f / DESR;
  for (int s = lane; s < PATCH; s += 32) {
    float x = dxp[s*3], y = dxp[s*3+1], z = dxp[s*3+2];
    dxp[s*3+0] = (R00*x + R01*y + R02*z) * ir;
    dxp[s*3+1] = (R10*x + R11*y + R12*z) * ir;
    dxp[s*3+2] = (R20*x + R21*y + R22*z) * ir;
  }
}

// ---------------------------------------------------------------------------
// Kernel 3: fused voxel grouping + azimuth rotation + PointNet (16ch) + max.
// Patch staged into LDS via the CDNA5 async global->LDS copy engine.
// ---------------------------------------------------------------------------
__global__ __launch_bounds__(256)
void k_voxel_pnt(const float* __restrict__ patch,
                 const float* __restrict__ W, const float* __restrict__ bias,
                 const float* __restrict__ bng, const float* __restrict__ bnb,
                 const float* __restrict__ bnm, const float* __restrict__ bnv,
                 float* __restrict__ x0) {
  __shared__ __attribute__((aligned(16))) float sp[PATCH*3];
  __shared__ float sg[8][VOXS][3];
  __shared__ float swv[16*3];
  __shared__ float ssc[16], ssh[16];
  int m = blockIdx.x, tid = threadIdx.x;
  // Async global->LDS: 6144B contiguous, 16B per lane, ASYNCcnt-tracked.
  {
    unsigned ldsbase = (unsigned)(unsigned long long)(size_t)&sp[0];
    const char* gp = (const char*)(patch + (size_t)m * PATCH * 3);
    for (int off = tid * 16; off < PATCH*3*4; off += 256*16) {
      unsigned laddr = ldsbase + (unsigned)off;
      unsigned long long ga = (unsigned long long)(size_t)(gp + off);
      asm volatile("global_load_async_to_lds_b128 %0, %1, off"
                   :: "v"(laddr), "v"(ga) : "memory");
    }
    asm volatile("s_wait_asynccnt 0" ::: "memory");
  }
  if (tid < 48) swv[tid] = W[tid];
  if (tid < 16) {
    float sc = bng[tid] * rsqrtf(bnv[tid] + BN_EPS);
    ssc[tid] = sc;
    ssh[tid] = sc * (bias[tid] - bnm[tid]) + bnb[tid];
  }
  __syncthreads();
  int wave = tid >> 5, lane = tid & 31;
  for (int vb = 0; vb < NVOX; vb += 8) {
    int v = vb + wave;
    bool act = v < NVOX;
    if (act) {
      int ir = v / NPOS2, ie = (v / AZIN) % ELEN, ia = v % AZIN;
      float rr = (ir + 0.5f) / (float)RADN;
      float th = (ie + 0.5f) * PI_F / (float)ELEN;
      float ph = (ia + 0.5f) * 2.f * PI_F / (float)AZIN;
      float st = sinf(th);
      float cx = rr*st*cosf(ph), cy = rr*st*sinf(ph), cz = rr*cosf(th);
      float ang = -2.f * PI_F * (float)ia / (float)AZIN;
      float ca = cosf(ang), sa = sinf(ang);
      int cnt = 0;
      for (int base = 0; base < PATCH; base += 32) {
        if (cnt >= VOXS) break;
        int s = base + lane;
        float px = sp[s*3], py = sp[s*3+1], pz = sp[s*3+2];
        float ax = px-cx, ay = py-cy, az = pz-cz;
        bool in = (ax*ax + ay*ay + az*az) < (VRAD*VRAD);
        unsigned mask = (unsigned)__ballot(in);
        int pre = __popc(mask & ((1u << lane) - 1u));
        if (in && (cnt + pre) < VOXS) {
          int n = cnt + pre;
          sg[wave][n][0] = ca*ax - sa*ay;
          sg[wave][n][1] = sa*ax + ca*ay;
          sg[wave][n][2] = az;
        }
        cnt += __popc(mask);
      }
      if (cnt > VOXS) cnt = VOXS;
      for (int n = lane; n < VOXS; n += 32)
        if (n >= cnt) { sg[wave][n][0]=0.f; sg[wave][n][1]=0.f; sg[wave][n][2]=0.f; }
    }
    __syncthreads();
    if (act && lane < 16) {
      int c = lane;
      float wx = swv[c*3], wy = swv[c*3+1], wz = swv[c*3+2];
      float sc = ssc[c], sh = ssh[c];
      float mx = -3.4e38f;
      for (int n = 0; n < VOXS; ++n) {
        float val = sc*(wx*sg[wave][n][0] + wy*sg[wave][n][1] + wz*sg[wave][n][2]) + sh;
        val = val > 0.f ? val : 0.f;
        mx = val > mx ? val : mx;
      }
      x0[((size_t)m*16 + c)*NVOX + v] = mx;
    }
    __syncthreads();
  }
}

// ---------------------------------------------------------------------------
// Kernel 4: c3a — 3x3x3 conv 16->32 as WMMA GEMM. Activations staged into a
// zero/wrap-padded (5 x 9 x 22) x 16ch grid, channel-fastest, so each A/B
// fragment is two aligned ds_load_b128. K = 28 taps x 16ch (tap 27 zeroed).
// ---------------------------------------------------------------------------
#define P3R 5
#define P3E 9
#define P3A 22
__global__ __launch_bounds__(256)
void k_conv3a(const float* __restrict__ x0, const float* __restrict__ W,
              const float* __restrict__ bias,
              const float* __restrict__ bng, const float* __restrict__ bnb,
              const float* __restrict__ bnm, const float* __restrict__ bnv,
              float* __restrict__ x1) {
  __shared__ __attribute__((aligned(16))) _Float16 sxp[P3R*P3E*P3A][16]; // 31680B
  __shared__ __attribute__((aligned(16))) _Float16 sw[32][28*16];        // 28672B
  __shared__ float ssc[32], ssh[32];
  int m = blockIdx.x, tid = threadIdx.x;
  for (int i = tid; i < P3R*P3E*P3A*16; i += 256) {
    int pos = i >> 4, ch = i & 15;
    int r2 = pos / (P3E*P3A), rem = pos % (P3E*P3A), e2 = rem / P3A, a2 = rem % P3A;
    float val = 0.f;
    if (r2 >= 1 && r2 <= RADN && e2 >= 1 && e2 <= ELEN) {
      int a = (a2 - 1 + AZIN) % AZIN;
      val = x0[((size_t)m*16 + ch)*NVOX + ((r2-1)*ELEN + (e2-1))*AZIN + a];
    }
    sxp[pos][ch] = (_Float16)val;
  }
  for (int i = tid; i < 32*28*16; i += 256) {
    int o = i / 448, q = i % 448, t = q >> 4, ch = q & 15;
    sw[o][q] = (_Float16)(t < 27 ? W[o*432 + ch*27 + t] : 0.f);
  }
  if (tid < 32) {
    float sc = bng[tid] * rsqrtf(bnv[tid] + BN_EPS);
    ssc[tid] = sc;
    ssh[tid] = sc * (bias[tid] - bnm[tid]) + bnb[tid];
  }
  if (tid == 0) __builtin_prefetch(x0 + (((size_t)(m+1)&511)*16)*NVOX, 0, 3);
  __syncthreads();
  int wave = tid >> 5, lane = tid & 31, half = lane >> 4, rc = lane & 15;
  int chb = half * 8;
  for (int job = wave; job < 54; job += 8) {          // 27 M-tiles x 2 N-tiles
    int mt = job >> 1, nt = job & 1;
    int pc = imin(mt*16 + rc, NVOX-1);
    int ir = pc / NPOS2, ie = (pc / AZIN) % ELEN, ia = pc % AZIN;
    int pbase = ((ir+1)*P3E + (ie+1))*P3A + (ia+1);   // padded grid center
    int o = nt*16 + rc;
    V8F acc;
    #pragma unroll
    for (int r = 0; r < 8; ++r) acc.f[r] = 0.f;
    #pragma unroll 2
    for (int ks = 0; ks < 14; ++ks) {
      int t0 = 2*ks, t1 = 2*ks + 1;
      int dr0 = t0/9 - 1, de0 = (t0/3)%3 - 1, da0 = t0%3 - 1;
      int t1a = t1 < 27 ? t1 : 0;                     // A tap (B row is zero)
      int dr1 = t1a/9 - 1, de1 = (t1a/3)%3 - 1, da1 = t1a%3 - 1;
      int pos0 = pbase + (dr0*P3E + de0)*P3A + da0;
      int pos1 = pbase + (dr1*P3E + de1)*P3A + da1;
      V16H A, B;
      A.r[0] = *(const v8h*)&sxp[pos0][chb];
      A.r[1] = *(const v8h*)&sxp[pos1][chb];
      B.r[0] = *(const v8h*)&sw[o][t0*16 + chb];
      B.r[1] = *(const v8h*)&sw[o][t1*16 + chb];
      acc.v = __builtin_amdgcn_wmma_f32_16x16x32_f16(
                  false, A.v, false, B.v, (short)0, acc.v, false, false);
    }
    float sc = ssc[o], sh = ssh[o];
    #pragma unroll
    for (int r = 0; r < 8; ++r) {
      float val = sc*acc.f[r] + sh;
      acc.f[r] = val > 0.f ? val : 0.f;
    }
    int pb = mt*16 + half*8;
    float* dst = &x1[((size_t)m*32 + o)*NVOX + pb];
    if (mt != 26) {                                   // full tile: 2x b128
      *(float4*)(dst)     = make_float4(acc.f[0], acc.f[1], acc.f[2], acc.f[3]);
      *(float4*)(dst + 4) = make_float4(acc.f[4], acc.f[5], acc.f[6], acc.f[7]);
    } else {
      #pragma unroll
      for (int r = 0; r < 8; ++r) if (pb + r < NVOX) dst[r] = acc.f[r];
    }
  }
}

// ---------------------------------------------------------------------------
// Kernel 5: c3b — (3,1,1) conv 32->32 reducing the radial dim. K = 96.
// ---------------------------------------------------------------------------
__global__ __launch_bounds__(256)
void k_conv3b(const float* __restrict__ x1, const float* __restrict__ W,
              const float* __restrict__ bias,
              const float* __restrict__ bng, const float* __restrict__ bnb,
              const float* __restrict__ bnm, const float* __restrict__ bnv,
              float* __restrict__ x2) {
  __shared__ __attribute__((aligned(16))) _Float16 sx[RADN*NPOS2][32]; // 26880B
  __shared__ __attribute__((aligned(16))) _Float16 sw[32][96];         //  6144B
  __shared__ float ssc[32], ssh[32];
  int m = blockIdx.x, tid = threadIdx.x;
  for (int i = tid; i < RADN*NPOS2*32; i += 256) {
    int pos = i >> 5, ch = i & 31;                    // pos = r*140 + pos2d
    sx[pos][ch] = (_Float16)x1[((size_t)m*32 + ch)*NVOX + pos];
  }
  for (int i = tid; i < 32*96; i += 256) {
    int o = i / 96, q = i % 96, r = q >> 5, ii = q & 31;
    sw[o][q] = (_Float16)W[o*96 + ii*3 + r];
  }
  if (tid < 32) {
    float sc = bng[tid] * rsqrtf(bnv[tid] + BN_EPS);
    ssc[tid] = sc;
    ssh[tid] = sc * (bias[tid] - bnm[tid]) + bnb[tid];
  }
  __syncthreads();
  int wave = tid >> 5, lane = tid & 31, half = lane >> 4, rc = lane & 15;
  int chb = half * 8;
  for (int job = wave; job < 18; job += 8) {          // 9 M-tiles x 2 N-tiles
    int mt = job >> 1, nt = job & 1;
    int pc = imin(mt*16 + rc, NPOS2-1);
    int o = nt*16 + rc;
    V8F acc;
    #pragma unroll
    for (int r = 0; r < 8; ++r) acc.f[r] = 0.f;
    #pragma unroll
    for (int ks = 0; ks < 3; ++ks) {                  // radial tap
      V16H A, B;
      A.r[0] = *(const v8h*)&sx[ks*NPOS2 + pc][chb];
      A.r[1] = *(const v8h*)&sx[ks*NPOS2 + pc][16 + chb];
      B.r[0] = *(const v8h*)&sw[o][ks*32 + chb];
      B.r[1] = *(const v8h*)&sw[o][ks*32 + 16 + chb];
      acc.v = __builtin_amdgcn_wmma_f32_16x16x32_f16(
                  false, A.v, false, B.v, (short)0, acc.v, false, false);
    }
    float sc = ssc[o], sh = ssh[o];
    #pragma unroll
    for (int r = 0; r < 8; ++r) {
      float val = sc*acc.f[r] + sh;
      acc.f[r] = val > 0.f ? val : 0.f;
    }
    int pb = mt*16 + half*8;
    float* dst = &x2[((size_t)m*32 + o)*NPOS2 + pb];
    if (mt != 8) {
      *(float4*)(dst)     = make_float4(acc.f[0], acc.f[1], acc.f[2], acc.f[3]);
      *(float4*)(dst + 4) = make_float4(acc.f[4], acc.f[5], acc.f[6], acc.f[7]);
    } else {
      #pragma unroll
      for (int r = 0; r < 8; ++r) if (pb + r < NPOS2) dst[r] = acc.f[r];
    }
  }
}

// ---------------------------------------------------------------------------
// Kernel 6: c2a/c2b — 3x3 conv 32->32 on (7,20), circular azimuth.
// ---------------------------------------------------------------------------
__global__ __launch_bounds__(256)
void k_conv2(const float* __restrict__ xin, const float* __restrict__ W,
             const float* __restrict__ bias,
             const float* __restrict__ bng, const float* __restrict__ bnb,
             const float* __restrict__ bnm, const float* __restrict__ bnv,
             float* __restrict__ xout) {
  __shared__ __attribute__((aligned(16))) _Float16 sx[P3E*P3A][32];  // 12672B
  __shared__ __attribute__((aligned(16))) _Float16 sw[32][9*32];     // 18432B
  __shared__ float ssc[32], ssh[32];
  int m = blockIdx.x, tid = threadIdx.x;
  for (int i = tid; i < P3E*P3A*32; i += 256) {
    int pos = i >> 5, ch = i & 31;
    int e2 = pos / P3A, a2 = pos % P3A;
    float val = 0.f;
    if (e2 >= 1 && e2 <= ELEN) {
      int a = (a2 - 1 + AZIN) % AZIN;
      val = xin[((size_t)m*32 + ch)*NPOS2 + (e2-1)*AZIN + a];
    }
    sx[pos][ch] = (_Float16)val;
  }
  for (int i = tid; i < 32*288; i += 256) {
    int o = i / 288, q = i % 288, t = q >> 5, ii = q & 31;
    sw[o][q] = (_Float16)W[o*288 + ii*9 + t];
  }
  if (tid < 32) {
    float sc = bng[tid] * rsqrtf(bnv[tid] + BN_EPS);
    ssc[tid] = sc;
    ssh[tid] = sc * (bias[tid] - bnm[tid]) + bnb[tid];
  }
  __syncthreads();
  int wave = tid >> 5, lane = tid & 31, half = lane >> 4, rc = lane & 15;
  int chb = half * 8;
  for (int job = wave; job < 18; job += 8) {
    int mt = job >> 1, nt = job & 1;
    int pc = imin(mt*16 + rc, NPOS2-1);
    int ie = pc / AZIN, ia = pc % AZIN;
    int pbase = (ie+1)*P3A + (ia+1);
    int o = nt*16 + rc;
    V8F acc;
    #pragma unroll
    for (int r = 0; r < 8; ++r) acc.f[r] = 0.f;
    #pragma unroll 3
    for (int ks = 0; ks < 9; ++ks) {                  // tap, K = 32 channels
      int de = ks/3 - 1, da = ks%3 - 1;
      int pos = pbase + de*P3A + da;
      V16H A, B;
      A.r[0] = *(const v8h*)&sx[pos][chb];
      A.r[1] = *(const v8h*)&sx[pos][16 + chb];
      B.r[0] = *(const v8h*)&sw[o][ks*32 + chb];
      B.r[1] = *(const v8h*)&sw[o][ks*32 + 16 + chb];
      acc.v = __builtin_amdgcn_wmma_f32_16x16x32_f16(
                  false, A.v, false, B.v, (short)0, acc.v, false, false);
    }
    float sc = ssc[o], sh = ssh[o];
    #pragma unroll
    for (int r = 0; r < 8; ++r) {
      float val = sc*acc.f[r] + sh;
      acc.f[r] = val > 0.f ? val : 0.f;
    }
    int pb = mt*16 + half*8;
    float* dst = &xout[((size_t)m*32 + o)*NPOS2 + pb];
    if (mt != 8) {
      *(float4*)(dst)     = make_float4(acc.f[0], acc.f[1], acc.f[2], acc.f[3]);
      *(float4*)(dst + 4) = make_float4(acc.f[4], acc.f[5], acc.f[6], acc.f[7]);
    } else {
      #pragma unroll
      for (int r = 0; r < 8; ++r) if (pb + r < NPOS2) dst[r] = acc.f[r];
    }
  }
}

// ---------------------------------------------------------------------------
// Kernel 7: pa (32->16, WMMA) + pb (16->1) + weighted mean + L2 normalize.
// ---------------------------------------------------------------------------
__global__ __launch_bounds__(256)
void k_final(const float* __restrict__ x4,
             const float* __restrict__ paW, const float* __restrict__ paB,
             const float* __restrict__ pag, const float* __restrict__ pab,
             const float* __restrict__ pam, const float* __restrict__ pav,
             const float* __restrict__ pbW, const float* __restrict__ pbB,
             const float* __restrict__ pbg, const float* __restrict__ pbb,
             const float* __restrict__ pbm, const float* __restrict__ pbv,
             float* __restrict__ out) {
  __shared__ __attribute__((aligned(16))) _Float16 sx[NPOS2][32];   // 8960B
  __shared__ __attribute__((aligned(16))) _Float16 sw[16][32];
  __shared__ float ssc[16], ssh[16];
  __shared__ float sw1[16][NPOS2];
  __shared__ float sw2[NPOS2];
  int m = blockIdx.x, tid = threadIdx.x;
  for (int i = tid; i < NPOS2*32; i += 256) {
    int pos = i >> 5, ch = i & 31;
    sx[pos][ch] = (_Float16)x4[((size_t)m*32 + ch)*NPOS2 + pos];
  }
  if (tid < 16*32) sw[tid >> 5][tid & 31] = (_Float16)paW[tid];
  if (tid < 16) {
    float sc = pag[tid] * rsqrtf(pav[tid] + BN_EPS);
    ssc[tid] = sc;
    ssh[tid] = sc * (paB[tid] - pam[tid]) + pab[tid];
  }
  __syncthreads();
  int wave = tid >> 5, lane = tid & 31, half = lane >> 4, rc = lane & 15;
  int chb = half * 8;
  for (int job = wave; job < 9; job += 8) {           // 9 M-tiles, N = 16
    int mt = job;
    int pc = imin(mt*16 + rc, NPOS2-1);
    int o = rc;
    V8F acc;
    #pragma unroll
    for (int r = 0; r < 8; ++r) acc.f[r] = 0.f;
    V16H A, B;
    A.r[0] = *(const v8h*)&sx[pc][chb];
    A.r[1] = *(const v8h*)&sx[pc][16 + chb];
    B.r[0] = *(const v8h*)&sw[o][chb];
    B.r[1] = *(const v8h*)&sw[o][16 + chb];
    acc.v = __builtin_amdgcn_wmma_f32_16x16x32_f16(
                false, A.v, false, B.v, (short)0, acc.v, false, false);
    float sc = ssc[o], sh = ssh[o];
    #pragma unroll
    for (int r = 0; r < 8; ++r) {
      int pp = mt*16 + r + half*8;
      if (pp < NPOS2) {
        float val = sc*acc.f[r] + sh;
        sw1[o][pp] = val > 0.f ? val : 0.f;
      }
    }
  }
  __syncthreads();
  float pbsc = pbg[0] * rsqrtf(pbv[0] + BN_EPS);
  float pbsh = pbsc * (pbB[0] - pbm[0]) + pbb[0];
  for (int p = tid; p < NPOS2; p += 256) {
    float a = 0.f;
    #pragma unroll
    for (int c = 0; c < 16; ++c) a += pbW[c] * sw1[c][p];
    float val = pbsc*a + pbsh;
    sw2[p] = val > 0.f ? val : 0.f;
  }
  __syncthreads();
  if (tid < 32) {
    int c = tid;
    float f = 0.f;
    for (int p = 0; p < NPOS2; ++p)
      f += x4[((size_t)m*32 + c)*NPOS2 + p] * sw2[p];
    f *= (1.f / (float)NPOS2);
    float ss = f * f;
    for (int off = 16; off; off >>= 1) ss += __shfl_xor(ss, off, 32);
    out[m*32 + c] = f / (sqrtf(ss) + 1e-12f);
  }
}

// ---------------------------------------------------------------------------
// Host launcher. Input order: pts, kpts, then params flattened depth-first
// with alphabetical keys (jax tree order): c2a,c2b,c3a,c3b,pa,pb,pnt; each
// conv = {W, b, bn.b, bn.g, bn.m, bn.v}.
// ---------------------------------------------------------------------------
extern "C" void kernel_launch(void* const* d_in, const int* in_sizes, int n_in,
                              void* d_out, int out_size, void* d_ws, size_t ws_size,
                              hipStream_t stream) {
#define F(i) ((const float*)d_in[(i)])
  const float* pts  = F(0);
  const float* kpts = F(1);
  // param bases: c2a=2, c2b=8, c3a=14, c3b=20, pa=26, pb=32, pnt=38
  // kernel BN arg order is (g, b, m, v) = (base+3, base+2, base+4, base+5)
#define CONVA(base) F(base), F(base+1), F(base+3), F(base+2), F(base+4), F(base+5)

  float* ws      = (float*)d_ws;
  float* patches = ws;                          // 512*512*3   =   786,432
  float* x0      = patches + 786432;            // 512*16*420  = 3,440,640
  float* x1      = x0 + 3440640;                // 512*32*420  = 6,881,280
  float* x2      = x1 + 6881280;                // 512*32*140  = 2,293,760
  float* x3      = x2 + 2293760;
  float* x4      = x3 + 2293760;
  float* fout    = (float*)d_out;               // 512*32

  k_group    <<<MKPT,  32, 0, stream>>>(pts, kpts, patches);
  k_frame    <<<MKPT,  32, 0, stream>>>(kpts, patches);
  k_voxel_pnt<<<MKPT, 256, 0, stream>>>(patches, CONVA(38), x0);   // pnt
  k_conv3a   <<<MKPT, 256, 0, stream>>>(x0, CONVA(14), x1);        // c3a
  k_conv3b   <<<MKPT, 256, 0, stream>>>(x1, CONVA(20), x2);        // c3b
  k_conv2    <<<MKPT, 256, 0, stream>>>(x2, CONVA(2),  x3);        // c2a
  k_conv2    <<<MKPT, 256, 0, stream>>>(x3, CONVA(8),  x4);        // c2b
  k_final    <<<MKPT, 256, 0, stream>>>(x4,
                 F(26), F(27), F(29), F(28), F(30), F(31),         // pa
                 F(32), F(33), F(35), F(34), F(36), F(37),         // pb
                 fout);
#undef CONVA
#undef F
}
